// DeformUnit_3676492006248
// MI455X (gfx1250) — matrix operand order repaired and checked by reading, not compiled, not run on hardware
//
#include <hip/hip_runtime.h>
#include <hip/hip_bf16.h>
#include <cstdint>

typedef __attribute__((ext_vector_type(16))) __bf16 v16bf;
typedef __attribute__((ext_vector_type(8)))  float  v8f;

#define NPTS    9
#define INC     128
#define OUTC    256
#define BDIM    8
#define HW      4096            // 64*64
#define M_TOTAL 32768           // 8*64*64 pixels
#define K_TOTAL 1152            // 9*128
#define BN_EPS  1e-5f

// ---------- fp32 -> bf16 (round to nearest even) ----------
__device__ __forceinline__ unsigned short f2bf(float f) {
  unsigned int u = __float_as_uint(f);
  unsigned int r = u + 0x7FFFu + ((u >> 16) & 1u);
  return (unsigned short)(r >> 16);
}

// ---------- CDNA5 async DMA: global -> LDS, 16B per lane, ASYNCcnt ----------
__device__ __forceinline__ void async_b128(const void* gaddr, const void* lds) {
  unsigned int l = (unsigned int)(uintptr_t)lds;     // low 32 bits = LDS byte addr
  asm volatile("global_load_async_to_lds_b128 %0, %1, off"
               :: "v"(l), "v"((unsigned long long)(uintptr_t)gaddr)
               : "memory");
}
__device__ __forceinline__ void wait_async0() {
  asm volatile("s_wait_asynccnt 0x0" ::: "memory");
}

// ---------- weight prep: B^T row-major [N][K], k = n*128 + c ----------
__global__ void prep_w_main(const float* __restrict__ w_conv,
                            unsigned short* __restrict__ gB2) {
  int i = blockIdx.x * blockDim.x + threadIdx.x;
  if (i >= OUTC * K_TOTAL) return;
  int oc = i / K_TOTAL, k = i % K_TOTAL;
  int n = k >> 7, c = k & 127;
  gB2[i] = f2bf(w_conv[(oc * INC + c) * NPTS + n]);
}

__global__ void prep_w_offmask(const float* __restrict__ w_off,
                               const float* __restrict__ w_mask,
                               unsigned short* __restrict__ gB1) {
  int i = blockIdx.x * blockDim.x + threadIdx.x;
  if (i >= 32 * K_TOTAL) return;
  int j = i / K_TOTAL, k = i % K_TOTAL;
  int n = k >> 7, c = k & 127;
  float v = 0.f;
  if (j < 18)      v = w_off [(j        * INC + c) * NPTS + n];
  else if (j < 27) v = w_mask[((j - 18) * INC + c) * NPTS + n];
  gB1[i] = f2bf(v);
}

__global__ void prep_bn(const float* __restrict__ g, const float* __restrict__ b,
                        const float* __restrict__ mean, const float* __restrict__ var,
                        float* __restrict__ scale, float* __restrict__ shift) {
  int i = threadIdx.x;
  float inv = rsqrtf(var[i] + BN_EPS);
  float s = g[i] * inv;
  scale[i] = s;
  shift[i] = b[i] - mean[i] * s;
}

// ---------- im2col for the 3x3 pad-1 offset/mask conv ----------
__global__ __launch_bounds__(256) void im2col_k(const float* __restrict__ x,
                                                unsigned short* __restrict__ A) {
  int i = blockIdx.x * 256 + threadIdx.x;      // 32768*9*64 items, w fastest
  int w     = i & 63;
  int i2    = i >> 6;
  int cpair = i2 & 63;
  int i3    = i2 >> 6;
  int tap   = i3 % 9;
  int bh    = i3 / 9;                          // b*64 + h
  int b = bh >> 6, h = bh & 63;
  int dx = tap / 3 - 1, dy = tap % 3 - 1;
  int hh = h + dx, ww = w + dy;
  int c = cpair * 2;
  float v0 = 0.f, v1 = 0.f;
  if ((unsigned)hh < 64u && (unsigned)ww < 64u) {
    const float* xp = x + (b * INC + c) * HW + hh * 64 + ww;
    v0 = xp[0];
    v1 = xp[HW];
  }
  int p = bh * 64 + w;
  unsigned int pk = (unsigned)f2bf(v0) | ((unsigned)f2bf(v1) << 16);
  *(unsigned int*)&A[p * K_TOTAL + tap * INC + c] = pk;
}

// ---------- bilinear deformable sampling -> A2 = x_off bf16 [M][K] ----------
__global__ __launch_bounds__(256) void sample_k(const float* __restrict__ x,
                                                const float* __restrict__ c1, // [M][32]
                                                const float* __restrict__ b_off,
                                                const float* __restrict__ b_mask,
                                                unsigned short* __restrict__ A) {
  __shared__ int   s_idx[NPTS][64][4];
  __shared__ float s_g  [NPTS][64][4];
  const int bh = blockIdx.x;                   // one (b,h) row, 64 pixels
  const int b = bh >> 6, h = bh & 63;
  const int tid = threadIdx.x;

  for (int it = tid; it < 64 * NPTS; it += 256) {
    int w = it & 63;
    int n = it >> 6;
    int p = bh * 64 + w;
    const float* cr = c1 + p * 32;
    float ox = cr[n]        + b_off[n];
    float oy = cr[NPTS + n] + b_off[NPTS + n];
    float ml = cr[18 + n]   + b_mask[n];
    float m  = 1.f / (1.f + __expf(-ml));
    float px = ox + (float)(n / 3 - 1) + (float)(h + 1);
    float py = oy + (float)(n % 3 - 1) + (float)(w + 1);
    float fx = floorf(px), fy = floorf(py);
    float qltx = fminf(fmaxf(fx,       0.f), 65.f);
    float qlty = fminf(fmaxf(fy,       0.f), 65.f);
    float qrbx = fminf(fmaxf(fx + 1.f, 0.f), 65.f);
    float qrby = fminf(fmaxf(fy + 1.f, 0.f), 65.f);
    float pxc  = fminf(fmaxf(px, 0.f), 65.f);
    float pyc  = fminf(fmaxf(py, 0.f), 65.f);
    float glt = (1.f + (qltx - pxc)) * (1.f + (qlty - pyc));
    float grb = (1.f - (qrbx - pxc)) * (1.f - (qrby - pyc));
    float glb = (1.f + (qltx - pxc)) * (1.f - (qrby - pyc));
    float grt = (1.f - (qrbx - pxc)) * (1.f + (qlty - pyc));
    int ilx = (int)qltx, ily = (int)qlty, irx = (int)qrbx, iry = (int)qrby;
    int   qx[4] = {ilx, irx, ilx, irx};
    int   qy[4] = {ily, iry, iry, ily};
    float gg[4] = {glt, grb, glb, grt};
    #pragma unroll
    for (int s = 0; s < 4; s++) {
      int xh = qx[s] - 1, xw = qy[s] - 1;      // un-pad
      bool ok = ((unsigned)xh < 64u) && ((unsigned)xw < 64u);
      s_idx[n][w][s] = ok ? (xh * 64 + xw) : 0;
      s_g  [n][w][s] = ok ? gg[s] * m : 0.f;
    }
  }
  __syncthreads();

  const int w  = tid & 63;
  const int cg = tid >> 6;                     // 0..3
  const int p  = bh * 64 + w;
  unsigned short* arow = A + p * K_TOTAL;
  for (int cb = 0; cb < 16; cb++) {
    int c = cb * 8 + cg * 2;
    const float* xb0 = x + (b * INC + c) * HW;
    const float* xb1 = xb0 + HW;
    for (int n = 0; n < NPTS; n++) {
      float a0 = 0.f, a1 = 0.f;
      #pragma unroll
      for (int s = 0; s < 4; s++) {
        int   idx = s_idx[n][w][s];
        float g   = s_g  [n][w][s];
        a0 += g * xb0[idx];
        a1 += g * xb1[idx];
      }
      unsigned int pk = (unsigned)f2bf(a0) | ((unsigned)f2bf(a1) << 16);
      *(unsigned int*)&arow[n * INC + c] = pk;
    }
  }
}

// ---------- bf16 WMMA GEMM with async-DMA double-buffered LDS ----------
// wave tile 32x32 (2x2 v_wmma_f32_16x16x32_bf16); block = WM x WN waves.
template<int WM, int WN, bool EPI>
__global__ __launch_bounds__(WM * WN * 32) void gemm_wmma(
    const unsigned short* __restrict__ gA,     // [M][K] bf16 row-major
    const unsigned short* __restrict__ gB,     // [N][K] bf16 row-major (B^T)
    float* __restrict__ gC,
    const float* __restrict__ scale, const float* __restrict__ shift,
    int ldc) {
  constexpr int BM = WM * 32;
  constexpr int BN = WN * 32;
  constexpr int NT = WM * WN * 32;
  constexpr int LDA = 40;                      // 32 + 8 pad (bank hygiene)
  constexpr int NA = (BM * 4) / NT;            // 16B chunks per thread (A tile)
  static_assert((BM * 4) % NT == 0, "A chunks divisible");
  __shared__ __align__(16) unsigned short As[2][BM * LDA];
  __shared__ __align__(16) unsigned short Bs[2][BN * LDA];

  const int tid  = threadIdx.x;
  const int lane = tid & 31;
  const int wave = tid >> 5;
  const int wm = wave % WM;
  const int wn = wave / WM;
  const int m0 = blockIdx.y * BM;
  const int n0 = blockIdx.x * BN;
  const int l15 = lane & 15;
  const int hi  = lane >> 4;
  const int a_c0 = hi ? 8 : 0;                 // ISA A frag: K {0-7,16-23}|{8-15,24-31}
  const int b_c0 = hi ? 8 : 0;                 // ISA B frag: K {0-7,16-23}|{8-15,24-31}

  v8f acc[2][2] = {};
  union Frag { uint4 u[2]; v16bf v; };

  // issue one K-tile worth of async global->LDS DMA into buffer `buf`
  auto issue = [&](int k0, int buf) {
    #pragma unroll
    for (int i = 0; i < NA; i++) {
      int idx = tid + i * NT;
      int row = idx >> 2, ch = idx & 3;
      async_b128(gA + (m0 + row) * K_TOTAL + k0 + ch * 8,
                 &As[buf][row * LDA + ch * 8]);
    }
    if (BN * 4 >= NT) {
      #pragma unroll
      for (int i = 0; i < (BN * 4) / NT; i++) {
        int idx = tid + i * NT;
        int row = idx >> 2, ch = idx & 3;
        async_b128(gB + (n0 + row) * K_TOTAL + k0 + ch * 8,
                   &Bs[buf][row * LDA + ch * 8]);
      }
    } else {
      if (tid < BN * 4) {
        int row = tid >> 2, ch = tid & 3;
        async_b128(gB + (n0 + row) * K_TOTAL + k0 + ch * 8,
                   &Bs[buf][row * LDA + ch * 8]);
      }
    }
  };

  issue(0, 0);
  int buf = 0;
  for (int k0 = 0; k0 < K_TOTAL; k0 += 32, buf ^= 1) {
    wait_async0();                 // my DMAs for this tile landed in LDS
    __syncthreads();               // everyone's landed; other buffer free (WAR safe)
    if (k0 + 32 < K_TOTAL) issue(k0 + 32, buf ^ 1);

    Frag a[2], bf[2];
    #pragma unroll
    for (int tm = 0; tm < 2; tm++) {
      int row = wm * 32 + tm * 16 + l15;
      a[tm].u[0] = *(const uint4*)&As[buf][row * LDA + a_c0];
      a[tm].u[1] = *(const uint4*)&As[buf][row * LDA + a_c0 + 16];
    }
    #pragma unroll
    for (int tn = 0; tn < 2; tn++) {
      int col = wn * 32 + tn * 16 + l15;
      bf[tn].u[0] = *(const uint4*)&Bs[buf][col * LDA + b_c0];
      bf[tn].u[1] = *(const uint4*)&Bs[buf][col * LDA + b_c0 + 16];
    }
    #pragma unroll
    for (int tm = 0; tm < 2; tm++)
      #pragma unroll
      for (int tn = 0; tn < 2; tn++)
        acc[tm][tn] = __builtin_amdgcn_wmma_f32_16x16x32_bf16(
            false, a[tm].v, false, bf[tn].v, (short)0, acc[tm][tn], false, false);
  }

  // epilogue: C tile layout (lanes 0-15: M=v, 16-31: M=v+8; N = lane&15)
  #pragma unroll
  for (int tm = 0; tm < 2; tm++) {
    #pragma unroll
    for (int tn = 0; tn < 2; tn++) {
      int col   = n0 + wn * 32 + tn * 16 + l15;
      int rbase = m0 + wm * 32 + tm * 16 + hi * 8;
      if (EPI) {                               // fused BN + ReLU, NCHW store
        float sc = scale[col], sh = shift[col];
        int bb = rbase >> 12, hw = rbase & 4095;   // 8 rows stay in one image
        float* dst = gC + ((bb << 8) + col) * HW + hw;
        float r[8];
        #pragma unroll
        for (int v = 0; v < 8; v++) {
          float val = acc[tm][tn][v] * sc + sh;
          r[v] = val > 0.f ? val : 0.f;
        }
        float4 q0 = {r[0], r[1], r[2], r[3]};
        float4 q1 = {r[4], r[5], r[6], r[7]};
        *(float4*)(dst)     = q0;
        *(float4*)(dst + 4) = q1;
      } else {
        #pragma unroll
        for (int v = 0; v < 8; v++)
          gC[(rbase + v) * ldc + col] = acc[tm][tn][v];
      }
    }
  }
}

extern "C" void kernel_launch(void* const* d_in, const int* in_sizes, int n_in,
                              void* d_out, int out_size, void* d_ws, size_t ws_size,
                              hipStream_t stream) {
  const float* x      = (const float*)d_in[0];
  const float* w_off  = (const float*)d_in[1];
  const float* b_off  = (const float*)d_in[2];
  const float* w_mask = (const float*)d_in[3];
  const float* b_mask = (const float*)d_in[4];
  const float* w_conv = (const float*)d_in[5];
  const float* bn_g   = (const float*)d_in[6];
  const float* bn_b   = (const float*)d_in[7];
  const float* bn_m   = (const float*)d_in[8];
  const float* bn_v   = (const float*)d_in[9];
  float* out = (float*)d_out;

  char* ws = (char*)d_ws;
  size_t szA  = (size_t)M_TOTAL * K_TOTAL * 2;   // 75.5 MB (A1, then reused as x_off)
  size_t szB2 = (size_t)OUTC * K_TOTAL * 2;
  size_t szB1 = (size_t)32 * K_TOTAL * 2;
  size_t szC1 = (size_t)M_TOTAL * 32 * 4;
  unsigned short* A   = (unsigned short*)(ws);
  unsigned short* gB2 = (unsigned short*)(ws + szA);
  unsigned short* gB1 = (unsigned short*)(ws + szA + szB2);
  float*          C1  = (float*)(ws + szA + szB2 + szB1);
  float*          sc  = (float*)(ws + szA + szB2 + szB1 + szC1);
  float*          sh  = sc + OUTC;

  prep_w_main   <<<(OUTC * K_TOTAL + 255) / 256, 256, 0, stream>>>(w_conv, gB2);
  prep_w_offmask<<<(32   * K_TOTAL + 255) / 256, 256, 0, stream>>>(w_off, w_mask, gB1);
  prep_bn       <<<1, 256, 0, stream>>>(bn_g, bn_b, bn_m, bn_v, sc, sh);

  // offset/mask conv as GEMM: [32768 x 1152] x [1152 x 32]
  im2col_k<<<(M_TOTAL * 9 * 64) / 256, 256, 0, stream>>>(x, A);
  gemm_wmma<8, 1, false><<<dim3(1, M_TOTAL / 256), 256, 0, stream>>>(
      A, gB1, C1, nullptr, nullptr, 32);

  // deformable bilinear sampling -> x_off bf16 (overwrites A workspace)
  sample_k<<<BDIM * 64, 256, 0, stream>>>(x, C1, b_off, b_mask, A);

  // main contraction + BN + ReLU: [32768 x 1152] x [1152 x 256] -> NCHW out
  gemm_wmma<4, 2, true><<<dim3(OUTC / 64, M_TOTAL / 128), 256, 0, stream>>>(
      A, gB2, out, sc, sh, 0);
}
